// LinearAttention_7000796692698
// MI455X (gfx1250) — matrix operand order repaired and checked by reading, not compiled, not run on hardware
//
#include <hip/hip_runtime.h>
#include <hip/hip_bf16.h>
#include <cstddef>
#include <cstdint>

// ---------------------------------------------------------------------------
// CDNA5 (gfx1250) wave32 WMMA bf16 GEMM pipeline for linear attention.
//   Q = elu(x Wq^T + bq)+1 ; K = elu(x Wk^T + bk)+1 ; V = x Wv^T + bv
//   KV[b] = K_b^T V_b ; O1 = Q_b KV[b] ; out = O1 Wo^T + bo
// v_wmma_f32_16x16x32_bf16 everywhere; contiguous staging uses the CDNA5
// async DMA path (global_load_async_to_lds_b128 + s_wait_asynccnt); LDS is
// double-buffered with one workgroup barrier per K-chunk; transposed fills
// stage k-pairs so the LDS scatter uses ds_store_b32; all dims compile-time.
// ---------------------------------------------------------------------------

typedef __attribute__((ext_vector_type(16))) __bf16 v16bf;
typedef __attribute__((ext_vector_type(8)))  __bf16 v8bf;
typedef __attribute__((ext_vector_type(4)))  __bf16 v4bf;
typedef __attribute__((ext_vector_type(2)))  __bf16 v2bf;
typedef __attribute__((ext_vector_type(8)))  float  v8f;

constexpr int S_DIM = 4096;
constexpr int B_DIM = 8;
constexpr int D_DIM = 1024;
constexpr int M_DIM = S_DIM * B_DIM;      // (S,B,D) == (S*B, D) row-major

#define BM   128
#define BN   64
#define BK   64
#define LDSK 72     // BK + 8 bf16 -> 144 B row stride (16B aligned, bank-skewed)
#define NACC (BN / 16)

// --- CDNA5 async memory->LDS copy (CDNA5 ISA, GV mode) ------
__device__ __forceinline__ void async_copy16(uint32_t lds_addr, const void* gptr) {
    asm volatile("global_load_async_to_lds_b128 %0, %1, off"
                 :: "v"(lds_addr), "v"(gptr)
                 : "memory");
}
__device__ __forceinline__ void async_wait0() {
    asm volatile("s_wait_asynccnt 0" ::: "memory");
}

// A-matrix 16x32 bf16 fragment, ISA 7.12.2 layout (two aligned b128 loads).
__device__ __forceinline__ v16bf frag_a(const __bf16* base, int lane, int ks) {
    const int row = lane & 15;
    const int kb  = (lane >> 4) * 8;
    const __bf16* p = base + row * LDSK + ks + kb;
    v8bf lo = *reinterpret_cast<const v8bf*>(p);
    v8bf hi = *reinterpret_cast<const v8bf*>(p + 16);
    return __builtin_shufflevector(lo, hi, 0, 1, 2, 3, 4, 5, 6, 7,
                                           8, 9, 10, 11, 12, 13, 14, 15);
}

// B-matrix 32x16 bf16 fragment: lane = column, K in ISA table order (matches frag_a).
__device__ __forceinline__ v16bf frag_b(const __bf16* base, int lane, int ks) {
    const int col = lane & 15;
    const int kb  = (lane >> 4) * 8;
    const __bf16* p = base + col * LDSK + ks + kb;
    v8bf lo = *reinterpret_cast<const v8bf*>(p);
    v8bf hi = *reinterpret_cast<const v8bf*>(p + 16);
    return __builtin_shufflevector(lo, hi, 0, 1, 2, 3, 4, 5, 6, 7,
                                           8, 9, 10, 11, 12, 13, 14, 15);
}

// 2 k-steps x 4 n-subtiles = 8 WMMAs per staged buffer; fragments preloaded
// into distinct registers so DS waits stagger.
__device__ __forceinline__ void mma_tile(const __bf16 (*Als)[LDSK],
                                         const __bf16 (*Bls)[LDSK],
                                         int wave, int lane, v8f* acc) {
#pragma unroll
    for (int ks = 0; ks < BK; ks += 32) {
        v16bf af = frag_a(&Als[wave * 16][0], lane, ks);
        v16bf b0 = frag_b(&Bls[0][0],  lane, ks);
        v16bf b1 = frag_b(&Bls[16][0], lane, ks);
        v16bf b2 = frag_b(&Bls[32][0], lane, ks);
        v16bf b3 = frag_b(&Bls[48][0], lane, ks);
        acc[0] = __builtin_amdgcn_wmma_f32_16x16x32_bf16(false, af, false, b0, (short)0, acc[0], false, false);
        acc[1] = __builtin_amdgcn_wmma_f32_16x16x32_bf16(false, af, false, b1, (short)0, acc[1], false, false);
        acc[2] = __builtin_amdgcn_wmma_f32_16x16x32_bf16(false, af, false, b2, (short)0, acc[2], false, false);
        acc[3] = __builtin_amdgcn_wmma_f32_16x16x32_bf16(false, af, false, b3, (short)0, acc[3], false, false);
    }
}

// k-contiguous bf16 tile (ROWS x BK) staged by async DMA, 16 B per op.
template <int ROWS, int LD>
__device__ __forceinline__ void stage_async(const __bf16* __restrict__ src,
                                            int row0, int kc, __bf16 (*lds)[LDSK], int tid) {
    const int f  = tid & 7;      // v8bf column within BK=64
    const int r0 = tid >> 3;     // 32 rows per pass
#pragma unroll
    for (int i = 0; i < ROWS / 32; ++i) {
        int r = r0 + i * 32;
        async_copy16((uint32_t)(uintptr_t)&lds[r][f * 8],
                     (const void*)&src[(size_t)(row0 + r) * LD + kc + f * 8]);
    }
}

// Transposed fill unit: read two adjacent k-rows (16 B each, coalesced) and
// scatter 8 packed bf16 pairs as ds_store_b32 into lds[c..c+7][kp..kp+1].
__device__ __forceinline__ void stage_tr_pair(const __bf16* __restrict__ row0,
                                              const __bf16* __restrict__ row1,
                                              __bf16 (*lds)[LDSK], int c, int kp) {
    v8bf v0 = *reinterpret_cast<const v8bf*>(row0);
    v8bf v1 = *reinterpret_cast<const v8bf*>(row1);
#pragma unroll
    for (int j = 0; j < 8; ++j) {
        v2bf t;
        t[0] = v0[j];
        t[1] = v1[j];
        *reinterpret_cast<v2bf*>(&lds[c + j][kp]) = t;   // 4 B, aligned (kp even)
    }
}

// ---------------------------------------------------------------------------
// fp32 -> bf16 streaming convert (x and the four weight matrices).
// ---------------------------------------------------------------------------
__global__ __launch_bounds__(256) void cvt_f32_bf16(const float* __restrict__ src,
                                                    __bf16* __restrict__ dst, int n4) {
    int i = blockIdx.x * 256 + threadIdx.x;
    const int stride = gridDim.x * 256;
    for (; i < n4; i += stride) {
        float4 v = reinterpret_cast<const float4*>(src)[i];
        v4bf b;
        b[0] = (__bf16)v.x; b[1] = (__bf16)v.y; b[2] = (__bf16)v.z; b[3] = (__bf16)v.w;
        reinterpret_cast<v4bf*>(dst)[i] = b;
    }
}

// ---------------------------------------------------------------------------
// C[m,n] = act( sum_k A[m,k] * W[n,k] + bias[n] ), A bf16 MxD, W bf16 DxD.
// ACT==1 -> elu(z)+1 = z>0 ? z+1 : exp(z).
// ---------------------------------------------------------------------------
template <int ACT>
__global__ __launch_bounds__(256) void gemm_xwT(
    const __bf16* __restrict__ A, const __bf16* __restrict__ W,
    const float* __restrict__ bias,
    __bf16* __restrict__ outBf, float* __restrict__ outF) {
    __shared__ __bf16 Als[2][BM][LDSK];
    __shared__ __bf16 Bls[2][BN][LDSK];

    const int tid  = threadIdx.x;
    const int lane = tid & 31;
    const int wave = tid >> 5;
    const int m0   = blockIdx.y * BM;
    const int n0   = blockIdx.x * BN;

    v8f acc[NACC] = {};

    constexpr int nCh = D_DIM / BK;
    stage_async<BM, D_DIM>(A, m0, 0, Als[0], tid);
    stage_async<BN, D_DIM>(W, n0, 0, Bls[0], tid);

    for (int c = 0; c < nCh; ++c) {
        async_wait0();
        __syncthreads();                       // buffer c ready for everyone
        const int cur = c & 1;
        if (c + 1 < nCh) {                     // prefetch next chunk into other buffer
            stage_async<BM, D_DIM>(A, m0, (c + 1) * BK, Als[cur ^ 1], tid);
            stage_async<BN, D_DIM>(W, n0, (c + 1) * BK, Bls[cur ^ 1], tid);
        }
        mma_tile(Als[cur], Bls[cur], wave, lane, acc);
    }

    // C layout: lane = N (mod 16), VGPR r = M offset (+8 for lanes 16-31).
    const int mbase = m0 + wave * 16 + ((lane >> 4) * 8);
    const int ncol  = lane & 15;
#pragma unroll
    for (int nt = 0; nt < NACC; ++nt) {
        int n = n0 + nt * 16 + ncol;
        float bv = bias ? bias[n] : 0.0f;
#pragma unroll
        for (int r = 0; r < 8; ++r) {
            float v = acc[nt][r] + bv;
            if (ACT) v = (v > 0.0f) ? (v + 1.0f) : __expf(v);
            size_t o = (size_t)(mbase + r) * D_DIM + n;
            if (outBf) outBf[o] = (__bf16)v;
            if (outF)  outF[o]  = v;
        }
    }
}

// ---------------------------------------------------------------------------
// KV[b][d][e] = sum_s K[s,b,d] * V[s,b,e]  (per-batch K^T V, K-dim = S)
// Transposed fills stage adjacent k-pairs -> ds_store_b32 scatter.
// ---------------------------------------------------------------------------
__global__ __launch_bounds__(256) void kv_kernel(
    const __bf16* __restrict__ Kbf, const __bf16* __restrict__ Vbf,
    __bf16* __restrict__ KVbf) {
    __shared__ __bf16 Als[2][BM][LDSK];
    __shared__ __bf16 Bls[2][BN][LDSK];

    const int tid  = threadIdx.x;
    const int lane = tid & 31;
    const int wave = tid >> 5;
    const int b  = blockIdx.z;
    const int d0 = blockIdx.y * BM;
    const int e0 = blockIdx.x * BN;
    constexpr int ROWSTR = B_DIM * D_DIM;          // seq-row stride (elements)
    const size_t bOff = (size_t)b * D_DIM;

    v8f acc[NACC] = {};

    auto stage = [&](int sc, __bf16 (*Abuf)[LDSK], __bf16 (*Bbuf)[LDSK]) {
        // A: Abuf[d][k] = K[(sc+k), b, d0+d]; 512 pair-units over 2 passes.
        {
            const int d = (tid & 15) * 8;          // 16 chunks cover 128 d
#pragma unroll
            for (int i = 0; i < 2; ++i) {
                int kp = ((tid >> 4) + i * 16) * 2;
                const __bf16* g = &Kbf[(size_t)(sc + kp) * ROWSTR + bOff + d0 + d];
                stage_tr_pair(g, g + ROWSTR, Abuf, d, kp);
            }
        }
        // B: Bbuf[e][k] = V[(sc+k), b, e0+e]; 256 pair-units in 1 pass.
        {
            const int e  = (tid & 7) * 8;          // 8 chunks cover 64 e
            const int kp = (tid >> 3) * 2;
            const __bf16* g = &Vbf[(size_t)(sc + kp) * ROWSTR + bOff + e0 + e];
            stage_tr_pair(g, g + ROWSTR, Bbuf, e, kp);
        }
    };

    constexpr int nCh = S_DIM / BK;
    stage(0, Als[0], Bls[0]);
    for (int c = 0; c < nCh; ++c) {
        __syncthreads();                       // chunk c staged (dscnt waited)
        const int cur = c & 1;
        if (c + 1 < nCh) stage((c + 1) * BK, Als[cur ^ 1], Bls[cur ^ 1]);
        mma_tile(Als[cur], Bls[cur], wave, lane, acc);
    }

    const int dbase = d0 + wave * 16 + ((lane >> 4) * 8);
    const int ecol  = lane & 15;
#pragma unroll
    for (int nt = 0; nt < NACC; ++nt) {
        int e = e0 + nt * 16 + ecol;
#pragma unroll
        for (int r = 0; r < 8; ++r)
            KVbf[(size_t)b * D_DIM * D_DIM + (size_t)(dbase + r) * D_DIM + e] =
                (__bf16)acc[nt][r];
    }
}

// ---------------------------------------------------------------------------
// O1[s,b,e] = sum_d Q[s,b,d] * KV[b][d][e]  (per-batch Q @ KV, K-dim = D)
// A staged async (k-contiguous), B staged transposed (paired); double-buffered.
// ---------------------------------------------------------------------------
__global__ __launch_bounds__(256) void qkv_kernel(
    const __bf16* __restrict__ Qbf, const __bf16* __restrict__ KVbf,
    __bf16* __restrict__ O1) {
    __shared__ __bf16 Als[2][BM][LDSK];
    __shared__ __bf16 Bls[2][BN][LDSK];

    const int tid  = threadIdx.x;
    const int lane = tid & 31;
    const int wave = tid >> 5;
    const int b  = blockIdx.z;
    const int s0 = blockIdx.y * BM;
    const int e0 = blockIdx.x * BN;
    const __bf16* Ab = Qbf + (size_t)b * D_DIM;   // row r -> Q[s0+r, b, :]

    v8f acc[NACC] = {};

    auto stageB = [&](int kc, __bf16 (*Bbuf)[LDSK]) {
        const int e  = (tid & 7) * 8;
        const int kp = (tid >> 3) * 2;
        const __bf16* g = &KVbf[(size_t)b * D_DIM * D_DIM +
                                (size_t)(kc + kp) * D_DIM + e0 + e];
        stage_tr_pair(g, g + D_DIM, Bbuf, e, kp);
    };

    constexpr int nCh = D_DIM / BK;
    stage_async<BM, B_DIM * D_DIM>(Ab, s0, 0, Als[0], tid);
    stageB(0, Bls[0]);
    for (int c = 0; c < nCh; ++c) {
        async_wait0();
        __syncthreads();
        const int cur = c & 1;
        if (c + 1 < nCh) {
            stage_async<BM, B_DIM * D_DIM>(Ab, s0, (c + 1) * BK, Als[cur ^ 1], tid);
            stageB((c + 1) * BK, Bls[cur ^ 1]);
        }
        mma_tile(Als[cur], Bls[cur], wave, lane, acc);
    }

    const int sbase = s0 + wave * 16 + ((lane >> 4) * 8);
    const int ecol  = lane & 15;
#pragma unroll
    for (int nt = 0; nt < NACC; ++nt) {
        int e = e0 + nt * 16 + ecol;
#pragma unroll
        for (int r = 0; r < 8; ++r)
            O1[((size_t)(sbase + r) * B_DIM + b) * D_DIM + e] = (__bf16)acc[nt][r];
    }
}

// ---------------------------------------------------------------------------
extern "C" void kernel_launch(void* const* d_in, const int* in_sizes, int n_in,
                              void* d_out, int out_size, void* d_ws, size_t ws_size,
                              hipStream_t stream) {
    (void)in_sizes; (void)n_in; (void)out_size; (void)ws_size;

    const float* x  = (const float*)d_in[0];
    const float* Wq = (const float*)d_in[1];
    const float* bq = (const float*)d_in[2];
    const float* Wk = (const float*)d_in[3];
    const float* bk = (const float*)d_in[4];
    const float* Wv = (const float*)d_in[5];
    const float* bv = (const float*)d_in[6];
    const float* Wo = (const float*)d_in[7];
    const float* bo = (const float*)d_in[8];

    // Workspace carve-out (all bf16):
    // xbf,Q,K,V,O1 = 64 MB each; KV = 16 MB; 4 weight copies = 2 MB each.
    __bf16* xbf = (__bf16*)d_ws;
    __bf16* Qbf = xbf + (size_t)M_DIM * D_DIM;
    __bf16* Kbf = Qbf + (size_t)M_DIM * D_DIM;
    __bf16* Vbf = Kbf + (size_t)M_DIM * D_DIM;
    __bf16* O1  = Vbf + (size_t)M_DIM * D_DIM;
    __bf16* KVb = O1  + (size_t)M_DIM * D_DIM;
    __bf16* Wqb = KVb + (size_t)B_DIM * D_DIM * D_DIM;
    __bf16* Wkb = Wqb + (size_t)D_DIM * D_DIM;
    __bf16* Wvb = Wkb + (size_t)D_DIM * D_DIM;
    __bf16* Wob = Wvb + (size_t)D_DIM * D_DIM;

    dim3 blk(256);

    // fp32 -> bf16 conversions (x once, weights once).
    {
        int n4x = (M_DIM * D_DIM) / 4;                  // 8M float4
        int n4w = (D_DIM * D_DIM) / 4;                  // 256K float4
        int gx = (n4x + 256 * 4 - 1) / (256 * 4);
        int gw = (n4w + 256 * 4 - 1) / (256 * 4);
        cvt_f32_bf16<<<gx, blk, 0, stream>>>(x,  xbf, n4x);
        cvt_f32_bf16<<<gw, blk, 0, stream>>>(Wq, Wqb, n4w);
        cvt_f32_bf16<<<gw, blk, 0, stream>>>(Wk, Wkb, n4w);
        cvt_f32_bf16<<<gw, blk, 0, stream>>>(Wv, Wvb, n4w);
        cvt_f32_bf16<<<gw, blk, 0, stream>>>(Wo, Wob, n4w);
    }

    dim3 gProj(D_DIM / BN, M_DIM / BM);   // 16 x 256 blocks

    gemm_xwT<1><<<gProj, blk, 0, stream>>>(xbf, Wqb, bq, Qbf, nullptr);
    gemm_xwT<1><<<gProj, blk, 0, stream>>>(xbf, Wkb, bk, Kbf, nullptr);
    gemm_xwT<0><<<gProj, blk, 0, stream>>>(xbf, Wvb, bv, Vbf, nullptr);

    kv_kernel<<<dim3(D_DIM / BN, D_DIM / BM, B_DIM), blk, 0, stream>>>(Kbf, Vbf, KVb);
    qkv_kernel<<<dim3(D_DIM / BN, S_DIM / BM, B_DIM), blk, 0, stream>>>(Qbf, KVb, O1);

    gemm_xwT<0><<<gProj, blk, 0, stream>>>(O1, Wob, bo, nullptr, (float*)d_out);
}
